// GATCONV_31602369364482
// MI455X (gfx1250) — compile-verified, matrix operand order checked by reading, not run on hardware
//
#include <hip/hip_runtime.h>
#include <hip/hip_bf16.h>

typedef __bf16 v16bf __attribute__((ext_vector_type(16)));
typedef float  v8f   __attribute__((ext_vector_type(8)));
typedef unsigned int v4u  __attribute__((ext_vector_type(4)));
typedef unsigned int u32x4 __attribute__((ext_vector_type(4)));
typedef int          i32x4 __attribute__((ext_vector_type(4)));
typedef int          i32x8 __attribute__((ext_vector_type(8)));

static constexpr int Nn   = 30000;
static constexpr int Ee   = 480000;
static constexpr int FIN  = 256;
static constexpr int DIMc = 32;
static constexpr int Hh   = 4;
static constexpr int HIDc = 128;
static constexpr int NCLS = 10;
static constexpr int EP   = Ee + Nn;   // edges + self loops
static constexpr float SLOPE = 0.2f;

union FragU { v4u u[2]; v16bf v; };

__device__ __forceinline__ unsigned short f2bf(float f) {
    unsigned u = __float_as_uint(f);
    unsigned r = u + 0x7FFFu + ((u >> 16) & 1u);   // round-to-nearest-even
    return (unsigned short)(r >> 16);
}

// ---------------------------------------------------------------------------
// TDM: issue a 2D bf16 tile load (tileRows x 32 elements, row stride = K elems)
// into LDS at ldsAddr. D# built per CDNA5 ISA §8 (group0/group1, 2D: g2=g3=0).
// Rows past tensorRows are OOB -> zero-filled by the TDM (handles M tail).
// ---------------------------------------------------------------------------
__device__ __forceinline__ void tdm_load_tile(unsigned ldsAddr,
                                              const unsigned short* gptr,
                                              int tensorRows, int K, int tileRows)
{
    unsigned long long ga = (unsigned long long)(size_t)gptr;
    u32x4 g0 = {
        1u,                                               // count=1 (valid), user mode
        ldsAddr,                                          // lds_addr (bytes)
        (unsigned)(ga & 0xFFFFFFFFu),                     // global_addr[31:0]
        (unsigned)((ga >> 32) & 0x1FFFFFFu) | (2u << 30)  // global_addr[56:32] | type=2
    };
    i32x8 g1 = {
        (int)(1u << 16),                                  // data_size=1 -> 2 bytes
        (int)((unsigned)K << 16),                         // tensor_dim0[15:0] (bits 63:48)
        (int)(((unsigned)tensorRows & 0xFFFFu) << 16),    // tensor_dim1[15:0] (bits 95:80)
        (int)(32u << 16),                                 // tile_dim0=32 (bits 127:112)
        (int)(unsigned)tileRows,                          // tile_dim1 (bits 143:128)
        (int)(unsigned)K,                                 // tensor_dim0_stride[31:0]
        0, 0
    };
    i32x4 gz = { 0, 0, 0, 0 };
#if __has_include(<hip/amd_detail/amd_gfx1250_TDM.h>)
    i32x8 gz8 = { 0, 0, 0, 0, 0, 0, 0, 0 };
    __builtin_amdgcn_tensor_load_to_lds(g0, g1, gz, gz, gz8, 0);   // 6-arg toolchain
#else
    __builtin_amdgcn_tensor_load_to_lds(g0, g1, gz, gz, 0);        // 5-arg ROCm 7.2
#endif
}

// ---------------------------------------------------------------------------
// f32 -> bf16 pack, 8 elements/thread (float4 x2 -> uint4)
// ---------------------------------------------------------------------------
__global__ __launch_bounds__(256)
void cvt_bf16(const float* __restrict__ src, unsigned short* __restrict__ dst, int n8)
{
    int i = blockIdx.x * blockDim.x + threadIdx.x;
    if (i >= n8) return;
    float4 a = ((const float4*)src)[i * 2];
    float4 b = ((const float4*)src)[i * 2 + 1];
    unsigned r0 = (unsigned)f2bf(a.x) | ((unsigned)f2bf(a.y) << 16);
    unsigned r1 = (unsigned)f2bf(a.z) | ((unsigned)f2bf(a.w) << 16);
    unsigned r2 = (unsigned)f2bf(b.x) | ((unsigned)f2bf(b.y) << 16);
    unsigned r3 = (unsigned)f2bf(b.z) | ((unsigned)f2bf(b.w) << 16);
    ((uint4*)dst)[i] = make_uint4(r0, r1, r2, r3);
}

// ---------------------------------------------------------------------------
// Y[M,Nout] = act(X[M,K] @ W[Nout,K]^T + bias), X/W pre-converted bf16.
// 256 threads = 8 waves; block = 128-row tile; wave = 16-row strip across Nout.
// A/B tiles double-buffered in LDS, filled by the Tensor Data Mover.
// ---------------------------------------------------------------------------
template<int NT, int ACT>
__global__ __launch_bounds__(256)
void gemm_wmma_tdm(const unsigned short* __restrict__ Xbf,
                   const unsigned short* __restrict__ Wbf,
                   const float* __restrict__ bias, float* __restrict__ Y,
                   int M, int K, int Nout)
{
    __shared__ v4u AsV[2][512];   // 2 x (128 rows x 32 k bf16) = 2 x 8 KB
    __shared__ v4u BsV[2][512];   // 2 x (Nout  x 32 k bf16)

    const int tid  = threadIdx.x;
    const int wave = tid >> 5;
    const int lane = tid & 31;
    const int lo16 = lane & 15;
    const int hi   = lane >> 4;
    const int rowBase = blockIdx.x * 128;
    const int rowsLeft = M - rowBase;           // TDM zero-fills rows beyond this
    const int S = K >> 5;                       // 32-wide K steps

    v8f acc[NT] = {};

    if (wave == 0) {
        tdm_load_tile((unsigned)(size_t)&AsV[0][0], Xbf + (size_t)rowBase * K,
                      rowsLeft, K, 128);
        tdm_load_tile((unsigned)(size_t)&BsV[0][0], Wbf,
                      Nout, K, Nout);
    }

    for (int s = 0; s < S; ++s) {
        const int buf = s & 1;
        if (wave == 0) {
            if (s + 1 < S) {
                int k0 = (s + 1) << 5;
                tdm_load_tile((unsigned)(size_t)&AsV[buf ^ 1][0],
                              Xbf + (size_t)rowBase * K + k0, rowsLeft, K, 128);
                tdm_load_tile((unsigned)(size_t)&BsV[buf ^ 1][0],
                              Wbf + k0, Nout, K, Nout);
                __builtin_amdgcn_s_wait_tensorcnt(2);   // current step's pair done
            } else {
                __builtin_amdgcn_s_wait_tensorcnt(0);
            }
        }
        __syncthreads();

        const v4u* A = AsV[buf];
        const v4u* B = BsV[buf];
        FragU a;
        int ar = wave * 16 + lo16;
        a.u[0] = A[ar * 4 + hi];
        a.u[1] = A[ar * 4 + hi + 2];
        #pragma unroll
        for (int t = 0; t < NT; ++t) {
            FragU b;
            int bn = t * 16 + lo16;
            b.u[0] = B[bn * 4 + hi];
            b.u[1] = B[bn * 4 + hi + 2];
            acc[t] = __builtin_amdgcn_wmma_f32_16x16x32_bf16(
                false, a.v, false, b.v, (short)0, acc[t], false, false);
        }
        __syncthreads();   // all waves done with buf before TDM may refill it
    }

    // C/D layout: lane -> col lo16; VGPR v -> row v + 8*hi
    #pragma unroll
    for (int t = 0; t < NT; ++t) {
        int col = t * 16 + lo16;
        float bcol = bias[col];
        #pragma unroll
        for (int v = 0; v < 8; ++v) {
            int m = rowBase + wave * 16 + v + 8 * hi;
            if (m < M) {
                float r = acc[t][v] + bcol;
                if (ACT) r = fmaxf(r, 0.f);
                Y[(size_t)m * Nout + col] = r;
            }
        }
    }
}

__global__ void init_layer(float* __restrict__ hnext, float* __restrict__ smax,
                           float* __restrict__ denom)
{
    int i = blockIdx.x * blockDim.x + threadIdx.x;
    if (i < Nn * HIDc) hnext[i] = 0.f;
    if (i < Nn * Hh) { smax[i] = -1.0e30f; denom[i] = 0.f; }
}

__device__ __forceinline__ void atomicMaxF(float* addr, float val) {
    if (val >= 0.f) atomicMax((int*)addr, __float_as_int(val));
    else            atomicMin((unsigned int*)addr, (unsigned int)__float_as_int(val));
}

// thread t -> (edge e = t/4, head h = t%4)
__global__ __launch_bounds__(256)
void edge_logits(const float* __restrict__ xl, const float* __restrict__ xr,
                 const int* __restrict__ src, const int* __restrict__ dst,
                 const float* __restrict__ att,
                 float* __restrict__ s, float* __restrict__ smax)
{
    __shared__ __align__(16) float attS[HIDc];
    if (threadIdx.x < HIDc) attS[threadIdx.x] = att[threadIdx.x];
    __syncthreads();
    int t = blockIdx.x * blockDim.x + threadIdx.x;
    if (t >= EP * Hh) return;
    int e = t >> 2, h = t & 3;
    int si, di;
    if (e < Ee) { si = src[e]; di = dst[e]; } else { si = di = e - Ee; }
    const float* pl = xl + (size_t)si * HIDc + h * DIMc;
    const float* pr = xr + (size_t)di * HIDc + h * DIMc;
    const float* pa = attS + h * DIMc;
    float acc = 0.f;
    #pragma unroll
    for (int c = 0; c < DIMc; c += 4) {
        float4 a = *(const float4*)(pl + c);
        float4 b = *(const float4*)(pr + c);
        float4 w = *(const float4*)(pa + c);
        float m;
        m = a.x + b.x; m = m > 0.f ? m : SLOPE * m; acc += m * w.x;
        m = a.y + b.y; m = m > 0.f ? m : SLOPE * m; acc += m * w.y;
        m = a.z + b.z; m = m > 0.f ? m : SLOPE * m; acc += m * w.z;
        m = a.w + b.w; m = m > 0.f ? m : SLOPE * m; acc += m * w.w;
    }
    s[t] = acc;
    atomicMaxF(smax + (size_t)di * Hh + h, acc);
}

__global__ __launch_bounds__(256)
void edge_p(float* __restrict__ s, const float* __restrict__ smax,
            const int* __restrict__ dst, float* __restrict__ denom)
{
    int t = blockIdx.x * blockDim.x + threadIdx.x;
    if (t >= EP * Hh) return;
    int e = t >> 2, h = t & 3;
    int di = (e < Ee) ? dst[e] : (e - Ee);
    float p = __expf(s[t] - smax[(size_t)di * Hh + h]);
    s[t] = p;
    atomicAdd(denom + (size_t)di * Hh + h, p);
}

__global__ __launch_bounds__(256)
void edge_agg(const float* __restrict__ p, const float* __restrict__ denom,
              const float* __restrict__ xl,
              const int* __restrict__ src, const int* __restrict__ dst,
              float* __restrict__ out)
{
    int t = blockIdx.x * blockDim.x + threadIdx.x;
    if (t >= EP * Hh) return;
    int e = t >> 2, h = t & 3;
    int si, di;
    if (e < Ee) { si = src[e]; di = dst[e]; } else { si = di = e - Ee; }
    float alpha = p[t] / denom[(size_t)di * Hh + h];
    const float* pl = xl + (size_t)si * HIDc + h * DIMc;
    float* po = out + (size_t)di * HIDc + h * DIMc;
    #pragma unroll
    for (int c = 0; c < DIMc; c += 4) {
        float4 a = *(const float4*)(pl + c);
        atomicAdd(po + c + 0, alpha * a.x);
        atomicAdd(po + c + 1, alpha * a.y);
        atomicAdd(po + c + 2, alpha * a.z);
        atomicAdd(po + c + 3, alpha * a.w);
    }
}

__global__ __launch_bounds__(256)
void bias_relu(float* __restrict__ h, const float* __restrict__ bo)
{
    int i = blockIdx.x * blockDim.x + threadIdx.x;
    if (i < Nn * HIDc) h[i] = fmaxf(h[i] + bo[i & (HIDc - 1)], 0.f);
}

// per original edge: e_out = [h[src], h[dst]] (64 f32), out = W2 @ e + b2 (10 f32)
__global__ __launch_bounds__(256)
void edge_out_k(const float* __restrict__ hf, const int* __restrict__ src0,
                const int* __restrict__ dst0, const float* __restrict__ W2,
                const float* __restrict__ b2, float* __restrict__ out,
                float* __restrict__ eout)
{
    __shared__ __align__(16) float w2s[NCLS * 2 * DIMc];   // 640 floats
    __shared__ float b2s[NCLS];
    for (int i = threadIdx.x; i < NCLS * 2 * DIMc; i += 256) w2s[i] = W2[i];
    if (threadIdx.x < NCLS) b2s[threadIdx.x] = b2[threadIdx.x];
    __syncthreads();
    int e = blockIdx.x * blockDim.x + threadIdx.x;
    if (e >= Ee) return;
    int si = src0[e], di = dst0[e];
    float v[2 * DIMc];
    #pragma unroll
    for (int c = 0; c < DIMc; c += 4) {
        float4 a = *(const float4*)(hf + (size_t)si * DIMc + c);
        v[c + 0] = a.x; v[c + 1] = a.y; v[c + 2] = a.z; v[c + 3] = a.w;
        float4 b = *(const float4*)(hf + (size_t)di * DIMc + c);
        v[DIMc + c + 0] = b.x; v[DIMc + c + 1] = b.y;
        v[DIMc + c + 2] = b.z; v[DIMc + c + 3] = b.w;
    }
    float* pe = eout + (size_t)e * (2 * DIMc);
    #pragma unroll
    for (int k = 0; k < 2 * DIMc; k += 4)
        *(float4*)(pe + k) = make_float4(v[k], v[k + 1], v[k + 2], v[k + 3]);
    #pragma unroll
    for (int j = 0; j < NCLS; ++j) {
        float acc = b2s[j];
        const float* wr = w2s + j * 2 * DIMc;
        #pragma unroll
        for (int k = 0; k < 2 * DIMc; ++k) acc += wr[k] * v[k];
        out[(size_t)e * NCLS + j] = acc;
    }
}

extern "C" void kernel_launch(void* const* d_in, const int* in_sizes, int n_in,
                              void* d_out, int out_size, void* d_ws, size_t ws_size,
                              hipStream_t stream)
{
    (void)in_sizes; (void)n_in; (void)out_size; (void)ws_size;
    const float* x    = (const float*)d_in[0];
    const int*   eidx = (const int*)d_in[1];
    const int*   src  = eidx;            // edge_index[0]
    const int*   dst  = eidx + Ee;       // edge_index[1]

    float* ws    = (float*)d_ws;
    float* xl    = ws;                               // N*128
    float* xr    = xl + (size_t)Nn * HIDc;           // N*128
    float* hA    = xr + (size_t)Nn * HIDc;           // N*128
    float* hB    = hA + (size_t)Nn * HIDc;           // N*128
    float* sP    = hB + (size_t)Nn * HIDc;           // EP*H
    float* smax  = sP + (size_t)EP * Hh;             // N*H
    float* denom = smax + (size_t)Nn * Hh;           // N*H
    float* hf    = denom + (size_t)Nn * Hh;          // N*32
    unsigned short* xbf = (unsigned short*)(hf + (size_t)Nn * DIMc);   // N*256 bf16
    unsigned short* wbf = xbf + (size_t)Nn * FIN;                      // all weights bf16

    dim3 blk(256);
    int gemmGrid = (Nn + 127) / 128;
    int egrid    = (EP * Hh + 255) / 256;
    int ngrid    = (Nn * HIDc + 255) / 256;

    // convert all weight matrices to bf16 once
    unsigned short* wl_bf[4]; unsigned short* wr_bf[4]; unsigned short* w1_bf;
    {
        size_t off = 0;
        for (int l = 0; l < 4; ++l) {
            int k = (l == 0) ? FIN : HIDc;
            int cnt = HIDc * k;
            wl_bf[l] = wbf + off; off += cnt;
            wr_bf[l] = wbf + off; off += cnt;
            cvt_bf16<<<(cnt / 8 + 255) / 256, blk, 0, stream>>>(
                (const float*)d_in[3 + l * 6 + 0], wl_bf[l], cnt / 8);
            cvt_bf16<<<(cnt / 8 + 255) / 256, blk, 0, stream>>>(
                (const float*)d_in[3 + l * 6 + 2], wr_bf[l], cnt / 8);
        }
        w1_bf = wbf + off;
        int cnt = DIMc * HIDc;
        cvt_bf16<<<(cnt / 8 + 255) / 256, blk, 0, stream>>>(
            (const float*)d_in[27], w1_bf, cnt / 8);
    }

    const float* feat = x;
    int K = FIN;
    float* hbuf[2] = { hA, hB };
    for (int l = 0; l < 4; ++l) {
        const float* bl  = (const float*)d_in[3 + l * 6 + 1];
        const float* br  = (const float*)d_in[3 + l * 6 + 3];
        const float* att = (const float*)d_in[3 + l * 6 + 4];
        const float* bo  = (const float*)d_in[3 + l * 6 + 5];
        float* hn = hbuf[l & 1];
        int n8 = Nn * K / 8;
        cvt_bf16<<<(n8 + 255) / 256, blk, 0, stream>>>(feat, xbf, n8);
        gemm_wmma_tdm<8, 0><<<gemmGrid, blk, 0, stream>>>(xbf, wl_bf[l], bl, xl, Nn, K, HIDc);
        gemm_wmma_tdm<8, 0><<<gemmGrid, blk, 0, stream>>>(xbf, wr_bf[l], br, xr, Nn, K, HIDc);
        init_layer<<<ngrid, blk, 0, stream>>>(hn, smax, denom);
        edge_logits<<<egrid, blk, 0, stream>>>(xl, xr, src, dst, att, sP, smax);
        edge_p<<<egrid, blk, 0, stream>>>(sP, smax, dst, denom);
        edge_agg<<<egrid, blk, 0, stream>>>(sP, denom, xl, src, dst, hn);
        bias_relu<<<ngrid, blk, 0, stream>>>(hn, bo);
        feat = hn;
        K = HIDc;
    }

    const float* b1 = (const float*)d_in[28];
    const float* W2 = (const float*)d_in[29];
    const float* b2 = (const float*)d_in[30];
    int n8 = Nn * HIDc / 8;
    cvt_bf16<<<(n8 + 255) / 256, blk, 0, stream>>>(feat, xbf, n8);
    gemm_wmma_tdm<2, 1><<<gemmGrid, blk, 0, stream>>>(xbf, w1_bf, b1, hf, Nn, HIDc, DIMc);

    float* out  = (float*)d_out;                    // [E, 10]
    float* eout = out + (size_t)Ee * NCLS;          // [E, 64]
    edge_out_k<<<(Ee + 255) / 256, blk, 0, stream>>>(hf, src, dst, W2, b2, out, eout);
}
